// CausalDenseSelfAttention_19859928776993
// MI455X (gfx1250) — matrix operand order repaired and checked by reading, not compiled, not run on hardware
//
#include <hip/hip_runtime.h>

// ---------------------------------------------------------------------------
// CDNA5 (gfx1250) causal self-attention: bf16 WMMA everywhere, f32 accum.
// Tile staging uses GLOBAL_LOAD_ASYNC_TO_LDS (ASYNCcnt) when available.
// ---------------------------------------------------------------------------

typedef __attribute__((ext_vector_type(16))) __bf16 bf16x16;
typedef __attribute__((ext_vector_type(8)))  float  f32x8;
typedef __attribute__((vector_size(16)))     int    i32x4_t;

union FragA { bf16x16 v; unsigned int u[8]; unsigned short h[16]; };
union FragC { f32x8   v; float f[8]; };

#if defined(__gfx1250__) && __has_builtin(__builtin_amdgcn_global_load_async_to_lds_b128)
#define HAVE_ASYNC_LDS 1
#else
#define HAVE_ASYNC_LDS 0
#endif

#define GLOBAL_AS __attribute__((address_space(1)))
#define LOCAL_AS  __attribute__((address_space(3)))

// 16-byte global -> LDS copy (async DMA path on gfx1250, sync fallback)
__device__ __forceinline__ void copy16_g2l(unsigned short* lds_dst,
                                           const unsigned short* gsrc) {
#if HAVE_ASYNC_LDS
  __builtin_amdgcn_global_load_async_to_lds_b128(
      (GLOBAL_AS i32x4_t*)gsrc,
      (LOCAL_AS  i32x4_t*)lds_dst, 0, 0);
#else
  *(uint4*)lds_dst = *(const uint4*)gsrc;
#endif
}

__device__ __forceinline__ void async_copy_fence() {
#if HAVE_ASYNC_LDS
#if __has_builtin(__builtin_amdgcn_s_wait_asynccnt)
  __builtin_amdgcn_s_wait_asynccnt(0);
#else
  asm volatile("s_wait_asynccnt 0x0" ::: "memory");
#endif
#endif
}

__device__ __forceinline__ unsigned short f2bf(float x) {
  union { float f; unsigned int u; } c; c.f = x;
  unsigned int r = c.u + 0x7FFFu + ((c.u >> 16) & 1u);   // round-nearest-even
  return (unsigned short)(r >> 16);
}

__device__ __forceinline__ void wmma_bf16(FragC& c, const FragA& a, const FragA& b) {
  c.v = __builtin_amdgcn_wmma_f32_16x16x32_bf16(false, a.v, false, b.v,
                                                (short)0, c.v, false, false);
}

// fp32 -> bf16 conversion (grid-stride)
__global__ void cvt_bf16_kernel(const float* __restrict__ in,
                                unsigned short* __restrict__ out, int n) {
  int i = blockIdx.x * blockDim.x + threadIdx.x;
  int stride = gridDim.x * blockDim.x;
  for (; i < n; i += stride) out[i] = f2bf(in[i]);
}

// ---------------------------------------------------------------------------
// Tiled bf16 GEMM: C[M,N] = A[M,K] * W[K,N] + bias
// Block tile 128x128, K-step 32. 8 waves: wm(0..3) x wn(0..1),
// each wave owns 2x4 = 8 16x16 f32 accumulators.
// MODE 0: QKV epilogue -> scatter bf16 into q/k/v [B,H,T,D]
// MODE 1: fp32 output  -> out[M,N]
// ---------------------------------------------------------------------------
template<int MODE>
__global__ __launch_bounds__(256)
void gemm_bf16_kernel(const unsigned short* __restrict__ A,
                      const unsigned short* __restrict__ W,
                      const float* __restrict__ bias,
                      int M, int N, int K,
                      unsigned short* __restrict__ qO,
                      unsigned short* __restrict__ kO,
                      unsigned short* __restrict__ vO,
                      float* __restrict__ out)
{
  constexpr int LDA = 40;   // 32 + pad (80B rows: 16B-aligned 16B chunks)
  constexpr int LDB = 136;  // 128 + pad (272B rows)
  __shared__ __align__(16) unsigned short sA[128 * LDA];
  __shared__ __align__(16) unsigned short sB[32 * LDB];

  const int tid  = threadIdx.x;
  const int lane = tid & 31;
  const int wave = tid >> 5;
  const int half = lane >> 4;
  const int l16  = lane & 15;
  const int wm   = wave >> 1;          // 0..3 -> 32 rows
  const int wn   = wave & 1;           // 0..1 -> 64 cols
  const int mBlk = blockIdx.y * 128;
  const int nBlk = blockIdx.x * 128;

  FragC c[2][4];
#pragma unroll
  for (int i = 0; i < 2; ++i)
#pragma unroll
    for (int j = 0; j < 4; ++j)
#pragma unroll
      for (int r = 0; r < 8; ++r) c[i][j].f[r] = 0.0f;

  for (int kt = 0; kt < K; kt += 32) {
    // stage A(128x32) and W(32x128) tiles into LDS via async DMA
    int idx = tid;
#pragma unroll
    for (int r = 0; r < 2; ++r, idx += 256) {
      int row = idx >> 2, ch = idx & 3;                      // 4 x 16B per A row
      copy16_g2l(sA + row * LDA + ch * 8,
                 A + (size_t)(mBlk + row) * K + kt + ch * 8);
    }
    idx = tid;
#pragma unroll
    for (int r = 0; r < 2; ++r, idx += 256) {
      int row = idx >> 4, ch = idx & 15;                     // 16 x 16B per W row
      copy16_g2l(sB + row * LDB + ch * 8,
                 W + (size_t)(kt + row) * N + nBlk + ch * 8);
    }
    async_copy_fence();
    __syncthreads();

    if (kt + 32 < K) {  // hint next K-slab toward GL2 (global_prefetch_b8)
      __builtin_prefetch(A + (size_t)(mBlk + (tid >> 1)) * K + kt + 32, 0, 3);
      __builtin_prefetch(W + (size_t)(kt + 32 + (tid & 31)) * N + nBlk + (tid >> 5) * 16, 0, 3);
    }

    // A fragments (16x32 bf16): lane = row, K-pairs per VGPR with half-wave interleave
    FragA a[2], b[4];
#pragma unroll
    for (int i = 0; i < 2; ++i) {
      const int mrow = wm * 32 + i * 16 + l16;
#pragma unroll
      for (int p = 0; p < 8; ++p) {
        const int k0 = 2 * p + 8 * (half + (p >= 4 ? 1 : 0));
        a[i].u[p] = *(const unsigned int*)(sA + mrow * LDA + k0);
      }
    }
    // B fragments (32x16): VGPR p holds K=2p+8*(half+(p>=4))(,+1), lane = N
#pragma unroll
    for (int j = 0; j < 4; ++j) {
      const int ncol = wn * 64 + j * 16 + l16;
#pragma unroll
      for (int p = 0; p < 8; ++p) {
        const int k0 = 2 * p + 8 * (half + (p >= 4 ? 1 : 0));
        unsigned int lo = sB[k0 * LDB + ncol];
        unsigned int hi = sB[(k0 + 1) * LDB + ncol];
        b[j].u[p] = lo | (hi << 16);
      }
    }
#pragma unroll
    for (int i = 0; i < 2; ++i)
#pragma unroll
      for (int j = 0; j < 4; ++j)
        wmma_bf16(c[i][j], a[i], b[j]);
    __syncthreads();
  }

  // epilogue. C layout: VGPR r -> row (8*half + r), lane l16 -> col.
#pragma unroll
  for (int i = 0; i < 2; ++i) {
#pragma unroll
    for (int j = 0; j < 4; ++j) {
      const int nCol = nBlk + wn * 64 + j * 16 + l16;
      const float bv = bias[nCol];
#pragma unroll
      for (int r = 0; r < 8; ++r) {
        const int mRow = mBlk + wm * 32 + 8 * half + r + i * 16;
        const float val = c[i][j].f[r] + bv;
        if (MODE == 0) {
          // scatter into q/k/v as [B,H,T,D] bf16
          const int bb  = mRow >> 11;          // / 2048
          const int t   = mRow & 2047;
          const int sel = nCol >> 11;          // 0=q 1=k 2=v (uniform per block)
          const int cc  = nCol & 2047;
          const int hh  = cc >> 7, dd = cc & 127;
          const size_t dst = (((size_t)(bb * 16 + hh) * 2048) + t) * 128 + dd;
          unsigned short* dp = (sel == 0) ? qO : (sel == 1) ? kO : vO;
          dp[dst] = f2bf(val);
        } else {
          out[(size_t)mRow * N + nCol] = val;
        }
      }
    }
  }
}

// ---------------------------------------------------------------------------
// Flash attention, one (b,h) x 64-q-row block per workgroup.
// 8 waves: wm(0..3) -> 16 q rows each, wn(0..1) -> 64 of 128 d-cols each.
// kv tiles of 32 keys staged in LDS (async DMA); online softmax in f32.
// ---------------------------------------------------------------------------
__global__ __launch_bounds__(256)
void attn_kernel(const unsigned short* __restrict__ Q,
                 const unsigned short* __restrict__ Km,
                 const unsigned short* __restrict__ V,
                 unsigned short* __restrict__ Y)
{
  constexpr int T = 2048, D = 128, H = 16;
  constexpr int LD = 136;
  __shared__ __align__(16) unsigned short sK[32 * LD];
  __shared__ __align__(16) unsigned short sV[32 * LD];
  __shared__ __align__(16) unsigned short sP[8 * 16 * 32];   // per-wave P buffers

  const int tid  = threadIdx.x;
  const int lane = tid & 31;
  const int wave = tid >> 5;
  const int half = lane >> 4;
  const int l16  = lane & 15;
  const int wm   = wave >> 1;
  const int wn   = wave & 1;
  const int qBase = blockIdx.x * 64;
  const int bh    = blockIdx.y;

  const unsigned short* Qh = Q  + (size_t)bh * T * D;
  const unsigned short* Kh = Km + (size_t)bh * T * D;
  const unsigned short* Vh = V  + (size_t)bh * T * D;

  // preload Q fragments for all 4 d-steps straight from global (pairs are
  // d-contiguous in [T,D] row-major, so each pair is one aligned b32 load)
  FragA qf[4];
  const int qRowMine = qBase + wm * 16 + l16;
#pragma unroll
  for (int ds = 0; ds < 4; ++ds)
#pragma unroll
    for (int p = 0; p < 8; ++p) {
      const int k0 = 2 * p + 8 * (half + (p >= 4 ? 1 : 0));
      qf[ds].u[p] = *(const unsigned int*)(Qh + (size_t)qRowMine * D + ds * 32 + k0);
    }

  FragC o[4];
#pragma unroll
  for (int j = 0; j < 4; ++j)
#pragma unroll
    for (int r = 0; r < 8; ++r) o[j].f[r] = 0.0f;

  float mrow[8], lrow[8];
#pragma unroll
  for (int r = 0; r < 8; ++r) { mrow[r] = -1e30f; lrow[r] = 0.0f; }

  const float scale = 0.08838834764831845f;   // 1/sqrt(128)
  const int kvEnd = qBase + 64;               // causal bound for this block
  unsigned short* myP = sP + wave * 512;

  for (int kv = 0; kv < kvEnd; kv += 32) {
    __syncthreads();
    int idx = tid;
#pragma unroll
    for (int r = 0; r < 2; ++r, idx += 256) {
      const int row = idx >> 4, ch = idx & 15;
      copy16_g2l(sK + row * LD + ch * 8, Kh + (size_t)(kv + row) * D + ch * 8);
      copy16_g2l(sV + row * LD + ch * 8, Vh + (size_t)(kv + row) * D + ch * 8);
    }
    async_copy_fence();
    __syncthreads();

    // S = Q K^T  (two 16-key groups, 4 d-steps of 32)
    FragC s[2];
#pragma unroll
    for (int nf = 0; nf < 2; ++nf) {
#pragma unroll
      for (int r = 0; r < 8; ++r) s[nf].f[r] = 0.0f;
#pragma unroll
      for (int ds = 0; ds < 4; ++ds) {
        FragA bk;   // B element (k=d-offset, n=key) = K[key][d]; pairs d-contiguous
        const int krow = 16 * nf + l16;
#pragma unroll
        for (int p = 0; p < 8; ++p) {
          const int dcol = ds * 32 + 2 * p + 8 * (half + (p >= 4 ? 1 : 0));
          bk.u[p] = *(const unsigned int*)(sK + krow * LD + dcol);
        }
        wmma_bf16(s[nf], qf[ds], bk);
      }
    }

    // online softmax per q-row (rows striped: VGPR r -> row 8*half+r)
    float alpha[8];
#pragma unroll
    for (int r = 0; r < 8; ++r) {
      const int qr = qBase + wm * 16 + 8 * half + r;
      const int key0 = kv + l16;
      const int key1 = kv + 16 + l16;
      const bool ok0 = key0 <= qr, ok1 = key1 <= qr;
      const float s0 = ok0 ? s[0].f[r] * scale : -1e30f;
      const float s1 = ok1 ? s[1].f[r] * scale : -1e30f;
      float tmax = fmaxf(s0, s1);
#pragma unroll
      for (int off = 1; off < 16; off <<= 1)
        tmax = fmaxf(tmax, __shfl_xor(tmax, off, 32));
      const float mNew = fmaxf(mrow[r], tmax);
      const float a = __expf(mrow[r] - mNew);
      const float p0 = ok0 ? __expf(s0 - mNew) : 0.0f;
      const float p1 = ok1 ? __expf(s1 - mNew) : 0.0f;
      float rs = p0 + p1;
#pragma unroll
      for (int off = 1; off < 16; off <<= 1)
        rs += __shfl_xor(rs, off, 32);
      lrow[r] = lrow[r] * a + rs;
      mrow[r] = mNew;
      alpha[r] = a;
      myP[(8 * half + r) * 32 + l16]      = f2bf(p0);   // C-layout -> A-layout
      myP[(8 * half + r) * 32 + 16 + l16] = f2bf(p1);   // via per-wave LDS buffer
    }

    // reload P as a 16x32 A-fragment (same-wave LDS RAW: DS is in-order,
    // compiler inserts s_wait_dscnt for the read results)
    FragA pf;
#pragma unroll
    for (int p = 0; p < 8; ++p) {
      const int k0 = 2 * p + 8 * (half + (p >= 4 ? 1 : 0));
      pf.u[p] = *(const unsigned int*)(myP + l16 * 32 + k0);
    }

    // rescale O then O += P * V
#pragma unroll
    for (int j = 0; j < 4; ++j)
#pragma unroll
      for (int r = 0; r < 8; ++r) o[j].f[r] *= alpha[r];
#pragma unroll
    for (int j = 0; j < 4; ++j) {
      FragA bv;   // B element (k=key-offset, n=d-col) = V[kv+k][d]
      const int dc = wn * 64 + j * 16 + l16;
#pragma unroll
      for (int p = 0; p < 8; ++p) {
        const int k0 = 2 * p + 8 * (half + (p >= 4 ? 1 : 0));
        unsigned int lo = sV[k0 * LD + dc];
        unsigned int hi = sV[(k0 + 1) * LD + dc];
        bv.u[p] = lo | (hi << 16);
      }
      wmma_bf16(o[j], pf, bv);
    }
  }

  // normalize and write y as bf16 [B, T, C] (proj GEMM input)
  const int b = bh >> 4, h = bh & 15;
#pragma unroll
  for (int j = 0; j < 4; ++j) {
    const int dc = wn * 64 + j * 16 + l16;
#pragma unroll
    for (int r = 0; r < 8; ++r) {
      const int t = qBase + wm * 16 + 8 * half + r;
      const float val = o[j].f[r] / lrow[r];
      Y[((size_t)b * T + t) * (H * D) + h * D + dc] = f2bf(val);
    }
  }
}

// ---------------------------------------------------------------------------
extern "C" void kernel_launch(void* const* d_in, const int* in_sizes, int n_in,
                              void* d_out, int out_size, void* d_ws, size_t ws_size,
                              hipStream_t stream)
{
  constexpr int B = 2, T = 2048, C = 2048, H = 16, D = 128;
  const float* x      = (const float*)d_in[0];
  const float* w_qkv  = (const float*)d_in[1];
  const float* b_qkv  = (const float*)d_in[2];
  const float* w_proj = (const float*)d_in[3];
  const float* b_proj = (const float*)d_in[4];
  float* out = (float*)d_out;

  unsigned short* ws = (unsigned short*)d_ws;
  size_t off = 0;
  unsigned short* xb     = ws + off; off += (size_t)B * T * C;       // x bf16
  unsigned short* wqkvb  = ws + off; off += (size_t)C * 3 * C;       // w_qkv bf16
  unsigned short* wprojb = ws + off; off += (size_t)C * C;           // w_proj bf16
  unsigned short* qb     = ws + off; off += (size_t)B * H * T * D;   // q [B,H,T,D]
  unsigned short* kb     = ws + off; off += (size_t)B * H * T * D;
  unsigned short* vb     = ws + off; off += (size_t)B * H * T * D;
  unsigned short* yb     = ws + off; off += (size_t)B * T * C;       // attn out bf16
  (void)ws_size; (void)in_sizes; (void)n_in; (void)out_size;

  cvt_bf16_kernel<<<512, 256, 0, stream>>>(x,      xb,     B * T * C);
  cvt_bf16_kernel<<<512, 256, 0, stream>>>(w_qkv,  wqkvb,  C * 3 * C);
  cvt_bf16_kernel<<<512, 256, 0, stream>>>(w_proj, wprojb, C * C);

  // qkv = x @ w_qkv + b_qkv  -> scatter to q/k/v
  gemm_bf16_kernel<0><<<dim3(48, 32), 256, 0, stream>>>(
      xb, wqkvb, b_qkv, B * T, 3 * C, C, qb, kb, vb, nullptr);

  // flash attention
  attn_kernel<<<dim3(T / 64, B * H), 256, 0, stream>>>(qb, kb, vb, yb);

  // out = y @ w_proj + b_proj
  gemm_bf16_kernel<1><<<dim3(16, 32), 256, 0, stream>>>(
      yb, wprojb, b_proj, B * T, C, C, nullptr, nullptr, nullptr, out);
}
